// SemiFewShotSegPartGraph_86517821211098
// MI455X (gfx1250) — matrix operand order repaired and checked by reading, not compiled, not run on hardware
//
#include <hip/hip_runtime.h>
#include <hip/hip_bf16.h>

typedef _Float16 v16h __attribute__((ext_vector_type(16)));
typedef _Float16 v8h  __attribute__((ext_vector_type(8)));
typedef _Float16 v2h  __attribute__((ext_vector_type(2)));
typedef float    v8f  __attribute__((ext_vector_type(8)));

#define NIMG 16
#define CCH  2048
#define SEGS 128
#define HWPX 2809      // 53*53
#define SEGRES 417
#define FR   53
#define STR  40        // LDS tile row stride in halves (80 B; 16B-aligned chunks)
#define EPSV 1e-8f
#define SCAL 20.0f

#define USE_ASYNC_LDS 1   // global_load_async_to_lds_b128 for the A tile

static __device__ __forceinline__ void wait_asynccnt0() {
#if USE_ASYNC_LDS
#if __has_builtin(__builtin_amdgcn_s_wait_asynccnt)
    __builtin_amdgcn_s_wait_asynccnt(0);
#else
    asm volatile("s_wait_asynccnt 0x0" ::: "memory");
#endif
#endif
}

// ---------------------------------------------------------------------------
// Kernel A: resize segment map to 53x53, count pixels/segment, zero normsq.
// ---------------------------------------------------------------------------
__global__ __launch_bounds__(256) void seg_resize_kernel(
    const int* __restrict__ un_seg, int* __restrict__ seg_flat,
    float* __restrict__ inv_cnt, float* __restrict__ normsq)
{
    const int n = blockIdx.x;
    const int t = threadIdx.x;
    __shared__ int cnt[SEGS];
    if (t < SEGS) cnt[t] = 0;
    __syncthreads();
    const int* sb = un_seg + (size_t)n * SEGRES * SEGRES;
    for (int hw = t; hw < HWPX; hw += 256) {
        int h = hw / FR, w = hw - h * FR;
        int ih = (h * SEGRES) / FR;          // torch 'nearest' convention
        int iw = (w * SEGRES) / FR;
        int s = sb[ih * SEGRES + iw];
        seg_flat[n * HWPX + hw] = s;
        atomicAdd(&cnt[s], 1);
        normsq[n * HWPX + hw] = 0.0f;
    }
    __syncthreads();
    if (t < SEGS) inv_cnt[n * SEGS + t] = 1.0f / fmaxf((float)cnt[t], 1.0f);
}

// ---------------------------------------------------------------------------
// Kernel B: one streaming pass over feats. Block = (image n, 64-channel chunk).
// Builds segment sums (LDS ds_add_f32) and per-pixel sum of squares.
// sums layout: [n][s][c]. Loads are unconditional (clamped address) so they
// pipeline; the boundary select only zeroes the value fed to the atomics.
// ---------------------------------------------------------------------------
#define NI 11          // ceil(2809/256)
#define ASTR 129       // padded LDS stride (==1 mod 64): conflict-free transpose
__global__ __launch_bounds__(256) void segsum_kernel(
    const float* __restrict__ feats, const int* __restrict__ seg_flat,
    float* __restrict__ sums, float* __restrict__ normsq)
{
    const int n = blockIdx.x;
    const int c0 = blockIdx.y * 64;
    const int t = threadIdx.x;

    __shared__ int   ldsSeg[HWPX];
    __shared__ float ldsAcc[64 * ASTR];   // ~33 KB, [c_local][s], padded

    for (int i = t; i < 64 * ASTR; i += 256) ldsAcc[i] = 0.0f;
    for (int i = t; i < HWPX; i += 256) ldsSeg[i] = seg_flat[n * HWPX + i];
    __syncthreads();

    // per-thread pixel slots: hw = t + i*256, clamped for address validity
    int   hwc[NI];
    bool  ok[NI];
    int   segv[NI];
    float nrm[NI];
#pragma unroll
    for (int i = 0; i < NI; ++i) {
        int hw = t + (i << 8);
        ok[i]  = (hw < HWPX);
        hwc[i] = ok[i] ? hw : (HWPX - 1);
        nrm[i] = 0.0f;
    }
    __syncthreads();
#pragma unroll
    for (int i = 0; i < NI; ++i) segv[i] = ldsSeg[hwc[i]];

    const float* fbase = feats + ((size_t)n * CCH + c0) * HWPX;
    for (int cl = 0; cl < 64; ++cl) {
        const float* row = fbase + (size_t)cl * HWPX;
        float v[NI];
#pragma unroll
        for (int i = 0; i < NI; ++i) v[i] = row[hwc[i]];        // unconditional
#pragma unroll
        for (int i = 0; i < NI; ++i) {
            float x = ok[i] ? v[i] : 0.0f;
            nrm[i] += x * x;
            atomicAdd(&ldsAcc[cl * ASTR + segv[i]], x);         // +0 is harmless
        }
    }
    __syncthreads();

    // transposed write-out: consecutive threads -> consecutive c (coalesced)
    for (int i = t; i < 64 * SEGS; i += 256) {
        int s  = i >> 6;          // 0..127
        int cl = i & 63;          // 0..63
        sums[((size_t)n * SEGS + s) * CCH + c0 + cl] = ldsAcc[cl * ASTR + s];
    }
#pragma unroll
    for (int i = 0; i < NI; ++i) {
        if (ok[i]) atomicAdd(&normsq[n * HWPX + hwc[i]], nrm[i]);
    }
}

// ---------------------------------------------------------------------------
// Kernel C: finalize prototypes: p = (sum/cnt); normalize over c; emit f16
// in [n][s][c] layout (contiguous c) -- fully coalesced both directions.
// ---------------------------------------------------------------------------
__global__ __launch_bounds__(128) void proto_kernel(
    const float* __restrict__ sums, const float* __restrict__ inv_cnt,
    _Float16* __restrict__ pf16)
{
    const int n = blockIdx.x >> 7;
    const int s = blockIdx.x & (SEGS - 1);
    const int t = threadIdx.x;
    __shared__ float red[128];

    const float* sb = sums + ((size_t)n * SEGS + s) * CCH;
    const float ic = inv_cnt[n * SEGS + s];

    float acc = 0.0f;
    for (int c = t; c < CCH; c += 128) {
        float v = sb[c] * ic;
        acc += v * v;
    }
    red[t] = acc;
    __syncthreads();
    for (int o = 64; o > 0; o >>= 1) {
        if (t < o) red[t] += red[t + o];
        __syncthreads();
    }
    const float invn = 1.0f / fmaxf(sqrtf(red[0]), EPSV);
    _Float16* pb = pf16 + ((size_t)n * SEGS + s) * CCH;
    for (int c = t; c < CCH; c += 128) {
        pb[c] = (_Float16)(sb[c] * ic * invn);
    }
}

// ---------------------------------------------------------------------------
// Kernel D: WMMA GEMM  G[s,pix] = sum_c p_norm[s,c] * f[c,pix]  + max epilogue.
// Block = (image n, 128-pixel tile). 8 waves: 2 (s, 64 rows) x 4 (pix, 32 cols).
// Double-buffered LDS tiles, one barrier per k-step.
//   A tile: two global_load_async_to_lds_b128 per thread (async engine).
//   B tile: 16 unconditional clamped f32 loads (immediate offsets off one
//           per-thread pointer), cvt to f16, 8 packed ds_store_b32.
// Out-of-range pixels feed only discarded output columns -> no selects needed.
// ---------------------------------------------------------------------------
__global__ __launch_bounds__(256) void gemm_max_kernel(
    const float* __restrict__ feats, const _Float16* __restrict__ pf16,
    const float* __restrict__ normsq, float* __restrict__ out)
{
    __shared__ __attribute__((aligned(16))) _Float16 ldsA[2][SEGS * STR]; // [s][k]
    __shared__ __attribute__((aligned(16))) _Float16 ldsB[2][128 * STR];  // [pix][k]
    __shared__ unsigned ldsR[128];

    const int n    = blockIdx.x;
    const int pix0 = blockIdx.y * 128;
    const int t    = threadIdx.x;
    const int lane = t & 31;
    const int wid  = t >> 5;
    const int m2   = wid & 1;        // s-block of 64
    const int n2   = wid >> 1;       // pix-block of 32
    const int hi   = lane >> 4;      // lane half
    const int l16  = lane & 15;

    const float*    fbase = feats + (size_t)n * CCH * HWPX;
    const _Float16* pbase = pf16 + (size_t)n * CCH * SEGS;   // [s][c]

    // ---- A staging state: thread t owns (row ss, 32B chunk h16) ----
    const int ss  = t >> 1;
    const int h16 = (t & 1) << 4;
    unsigned aoff = (unsigned)(((unsigned)ss * CCH + h16) * 2);   // bytes; +64/step
    const unsigned long long pbase_u64 = (unsigned long long)(const void*)pbase;
    const unsigned ldsA0 = (unsigned)(size_t)&ldsA[0][ss * STR + h16];
    const unsigned ldsA1 = (unsigned)(size_t)&ldsA[1][ss * STR + h16];

    // ---- B staging state: pp = t&127 constant; k advances by 4 per iter ----
    const int ppB = t & 127;
    const int kbB = (t >> 7) << 1;                 // 0 or 2
    const int pixc = min(pix0 + ppB, HWPX - 1);    // clamped: garbage cols unused
    const float* bcur = fbase + (size_t)kbB * HWPX + pixc;   // +32*HWPX per step
    const int ldsBoff = ppB * STR + kbB;           // halves

    v8f acc[4][2];
#pragma unroll
    for (int mt = 0; mt < 4; ++mt)
#pragma unroll
        for (int nt = 0; nt < 2; ++nt)
#pragma unroll
            for (int r = 0; r < 8; ++r) acc[mt][nt][r] = 0.0f;

#define STAGE(BUF)                                                                \
    do {                                                                          \
        /* A: async 16B copy, contiguous row chunk */                             \
        unsigned laddr = (BUF) ? ldsA1 : ldsA0;                                   \
        STAGE_A_IMPL(laddr, aoff);                                                \
        STAGE_A_IMPL(laddr + 16, aoff + 16);                                      \
        /* B: 16 unconditional loads at immediate offsets, then cvt+store */      \
        float v0[8], v1[8];                                                       \
        _Pragma("unroll")                                                         \
        for (int i = 0; i < 8; ++i) {                                             \
            const float* p = bcur + (size_t)(4 * i) * HWPX;                       \
            v0[i] = p[0];                                                         \
            v1[i] = p[HWPX];                                                      \
        }                                                                         \
        _Pragma("unroll")                                                         \
        for (int i = 0; i < 8; ++i) {                                             \
            v2h pk; pk[0] = (_Float16)v0[i]; pk[1] = (_Float16)v1[i];             \
            *(v2h*)&ldsB[BUF][ldsBoff + 4 * i] = pk;                              \
        }                                                                         \
        aoff += 64;                                                               \
        bcur += (size_t)32 * HWPX;                                                \
    } while (0)

#if USE_ASYNC_LDS
#define STAGE_A_IMPL(laddr, goff)                                                 \
    asm volatile("global_load_async_to_lds_b128 %0, %1, %2"                       \
                 :: "v"(laddr), "v"(goff), "s"(pbase_u64) : "memory")
#else
#define STAGE_A_IMPL(laddr, goff)                                                 \
    do {                                                                          \
        v8h av = *(const v8h*)((const char*)pbase + (size_t)(goff));              \
        *(v8h*)(size_t)(laddr) = av;                                              \
    } while (0)
#endif

    // prologue: stage tile 0 into buffer 0
    STAGE(0);
    wait_asynccnt0();
    __syncthreads();

    const int NK = CCH / 32;   // 64 k-steps
    for (int kk = 0; kk < NK; ++kk) {
        const int cur = kk & 1;
        const bool more = (kk + 1 < NK);
        if (more) STAGE(cur ^ 1);

        // B fragments: 32 contiguous bytes at k-offset hi*16
        v16h bfr[2];
#pragma unroll
        for (int nt = 0; nt < 2; ++nt) {
            const _Float16* bp = &ldsB[cur][(n2 * 32 + nt * 16 + l16) * STR + hi * 16];
            *(v8h*)&bfr[nt]       = *(const v8h*)(bp);
            *((v8h*)&bfr[nt] + 1) = *(const v8h*)(bp + 8);
        }
        // A fragments: 32 contiguous bytes at k-offset hi*16
#pragma unroll
        for (int mt = 0; mt < 4; ++mt) {
            const _Float16* ap = &ldsA[cur][(m2 * 64 + mt * 16 + l16) * STR + hi * 16];
            v16h afr;
            *(v8h*)&afr       = *(const v8h*)(ap);
            *((v8h*)&afr + 1) = *(const v8h*)(ap + 8);
#pragma unroll
            for (int nt = 0; nt < 2; ++nt) {
                acc[mt][nt] = __builtin_amdgcn_wmma_f32_16x16x32_f16(
                    false, afr, false, bfr[nt], (short)0, acc[mt][nt], false, false);
            }
        }
        if (more) wait_asynccnt0();
        __syncthreads();
    }

    // ---- epilogue: max over s, then scale by 20 / max(||f||, eps) ----
    if (t < 128) ldsR[t] = 0u;
    __syncthreads();

#pragma unroll
    for (int nt = 0; nt < 2; ++nt) {
        float m = acc[0][nt][0];
#pragma unroll
        for (int mt = 0; mt < 4; ++mt)
#pragma unroll
            for (int r = 0; r < 8; ++r) m = fmaxf(m, acc[mt][nt][r]);
        unsigned b = __float_as_uint(m);
        unsigned key = (b & 0x80000000u) ? ~b : (b | 0x80000000u);  // monotone map
        atomicMax(&ldsR[n2 * 32 + nt * 16 + l16], key);
    }
    __syncthreads();

    if (t < 128) {
        int pix = pix0 + t;
        if (pix < HWPX) {
            unsigned key = ldsR[t];
            unsigned b = (key & 0x80000000u) ? (key & 0x7FFFFFFFu) : ~key;
            float g  = __uint_as_float(b);
            float ns = normsq[n * HWPX + pix];
            out[n * HWPX + pix] = g * (SCAL / fmaxf(sqrtf(ns), EPSV));
        }
    }
}

// ---------------------------------------------------------------------------
extern "C" void kernel_launch(void* const* d_in, const int* in_sizes, int n_in,
                              void* d_out, int out_size, void* d_ws, size_t ws_size,
                              hipStream_t stream) {
    const float* un_feats = (const float*)d_in[0];   // [16,2048,53,53]
    const int*   un_seg   = (const int*)d_in[1];     // [16,417,417]
    float*       out      = (float*)d_out;           // [16,53,53]

    char* w = (char*)d_ws;
    size_t o = 0;
    float*    sums    = (float*)(w + o);  o += (size_t)NIMG * CCH * SEGS * 4;   // 16.78 MB
    _Float16* pf16    = (_Float16*)(w + o); o += (size_t)NIMG * CCH * SEGS * 2; // 8.39 MB
    int*      segflat = (int*)(w + o);    o += (size_t)NIMG * HWPX * 4;
    float*    invcnt  = (float*)(w + o);  o += (size_t)NIMG * SEGS * 4;
    float*    normsq  = (float*)(w + o);  o += (size_t)NIMG * HWPX * 4;
    (void)ws_size; (void)in_sizes; (void)n_in; (void)out_size;

    seg_resize_kernel<<<NIMG, 256, 0, stream>>>(un_seg, segflat, invcnt, normsq);
    segsum_kernel<<<dim3(NIMG, CCH / 64), 256, 0, stream>>>(un_feats, segflat, sums, normsq);
    proto_kernel<<<NIMG * SEGS, 128, 0, stream>>>(sums, invcnt, pf16);
    gemm_max_kernel<<<dim3(NIMG, (HWPX + 127) / 128), 256, 0, stream>>>(un_feats, pf16, normsq, out);
}
